// AdaLNSelfAttn_RoPE_26929444946153
// MI455X (gfx1250) — compile-verified, matrix operand order checked by reading, not run on hardware
//
#include <hip/hip_runtime.h>

#define NB 4
#define NL 2048
#define NC 1024
#define NH 16
#define NHD 64
#define NM (NB*NL)                 // 8192 rows
#define LN_EPSF 1e-6f
#define MAX_SCALEF 4.6051701859880914f   // log(100)

typedef __attribute__((ext_vector_type(16))) __bf16 v16bf;
typedef __attribute__((ext_vector_type(8)))  __bf16 v8bf;
typedef __attribute__((ext_vector_type(8)))  float  v8f;

static __device__ __forceinline__ unsigned short f2bf(float f) {
  unsigned u = __builtin_bit_cast(unsigned, f);
  u += 0x7FFFu + ((u >> 16) & 1u);           // round-to-nearest-even
  return (unsigned short)(u >> 16);
}
static __device__ __forceinline__ v16bf cat16(v8bf lo, v8bf hi) {
  return __builtin_shufflevector(lo, hi, 0,1,2,3,4,5,6,7,8,9,10,11,12,13,14,15);
}
static __device__ __forceinline__ float gelu_t(float x) {
  return 0.5f * x * (1.f + tanhf(0.7978845608f * (x + 0.044715f * x * x * x)));
}
// lane swizzle via LDS-permute hw; pattern must be an ICE -> template parameter
template <int PAT>
static __device__ __forceinline__ float swz(float v) {
  return __builtin_bit_cast(float, __builtin_amdgcn_ds_swizzle(__builtin_bit_cast(int, v), PAT));
}
static __device__ __forceinline__ float red16_max(float v) {
  v = fmaxf(v, swz<0x041F>(v)); v = fmaxf(v, swz<0x081F>(v));
  v = fmaxf(v, swz<0x101F>(v)); v = fmaxf(v, swz<0x201F>(v));
  return v;
}
static __device__ __forceinline__ float red16_sum(float v) {
  v += swz<0x041F>(v); v += swz<0x081F>(v);
  v += swz<0x101F>(v); v += swz<0x201F>(v);
  return v;
}
#define WMMA_BF16(A_, B_, C_) \
  __builtin_amdgcn_wmma_f32_16x16x32_bf16(false, (A_), false, (B_), (short)0, (C_), false, false)

// ---------------- elementwise / small kernels ----------------

__global__ void k_cvt(const float* __restrict__ s, unsigned short* __restrict__ d, int n) {
  int i = blockIdx.x * 256 + threadIdx.x;
  if (i < n) d[i] = f2bf(s[i]);
}

__global__ void k_qkvbias(const float* __restrict__ qb, const float* __restrict__ vb,
                          float* __restrict__ o) {
  int i = blockIdx.x * 256 + threadIdx.x;
  if (i < 3 * NC) o[i] = (i < NC) ? qb[i] : (i < 2 * NC ? 0.f : vb[i - 2 * NC]);
}

// ada = silu(cond) @ W_ada^T + b_ada   (tiny: 4 x 6144)
__global__ __launch_bounds__(256)
void k_ada(const float* __restrict__ cond, const float* __restrict__ Wada,
           const float* __restrict__ bada, float* __restrict__ ada) {
  __shared__ float sc[NC];
  const int t = threadIdx.x;
  const int b = blockIdx.x / 24;
  const int n = (blockIdx.x % 24) * 256 + t;
  for (int i = t; i < NC; i += 256) {
    float cv = cond[b * NC + i];
    sc[i] = cv / (1.f + __expf(-cv));
  }
  __syncthreads();
  float acc = bada[n];
  const float* wr = &Wada[(size_t)n * NC];
  for (int k = 0; k < NC; ++k) acc += sc[k] * wr[k];
  ada[(size_t)b * (6 * NC) + n] = acc;
}

// h = LN(x) * (ada[si]+1) + ada[shi]  -> bf16
__global__ __launch_bounds__(256)
void k_ln_mod(const float* __restrict__ xin, const float* __restrict__ ada,
              int si, int shi, unsigned short* __restrict__ hout) {
  __shared__ float r1[256], r2[256];
  const int row = blockIdx.x, t = threadIdx.x, b = row >> 11;
  const size_t base = (size_t)row * NC;
  float xv[4], s1 = 0.f, s2 = 0.f;
#pragma unroll
  for (int i = 0; i < 4; ++i) {
    float v = xin[base + t + i * 256];
    xv[i] = v; s1 += v; s2 += v * v;
  }
  r1[t] = s1; r2[t] = s2; __syncthreads();
  for (int s = 128; s > 0; s >>= 1) {
    if (t < s) { r1[t] += r1[t + s]; r2[t] += r2[t + s]; }
    __syncthreads();
  }
  const float mean = r1[0] * (1.f / NC);
  const float var  = r2[0] * (1.f / NC) - mean * mean;
  const float rstd = rsqrtf(var + LN_EPSF);
#pragma unroll
  for (int i = 0; i < 4; ++i) {
    const int c = t + i * 256;
    const float sa = ada[((size_t)b * 6 + si) * NC + c];
    const float sh = ada[((size_t)b * 6 + shi) * NC + c];
    hout[base + c] = f2bf((xv[i] - mean) * rstd * (sa + 1.f) + sh);
  }
}

// out = x + y * ada[gi]
__global__ __launch_bounds__(256)
void k_resid(const float* __restrict__ xin, const float* __restrict__ y,
             const float* __restrict__ ada, int gi, float* __restrict__ out) {
  const size_t i = (size_t)blockIdx.x * 256 + threadIdx.x;
  const int c = (int)(i & (NC - 1));
  const int b = (int)(i >> 21);                 // / (L*C) = 2^21
  out[i] = xin[i] + y[i] * ada[((size_t)b * 6 + gi) * NC + c];
}

// q/k norm + per-head scale + RoPE, write q,k [B,H,L,64] and V transposed [B,H,64,L]
__global__ __launch_bounds__(64)
void k_qkv_prep(const float* __restrict__ qkv, const float* __restrict__ slog,
                const float* __restrict__ fcos, const float* __restrict__ fsin,
                unsigned short* __restrict__ qb, unsigned short* __restrict__ kb,
                unsigned short* __restrict__ vt) {
  __shared__ float red[64];
  const int bi = blockIdx.x;
  const int h = bi & (NH - 1);
  const int l = (bi >> 4) & (NL - 1);
  const int b = bi >> 15;
  const int d = threadIdx.x;
  const size_t base = (size_t)(b * NL + l) * (3 * NC);
  const float q = qkv[base + h * NHD + d];
  const float k = qkv[base + NC + h * NHD + d];
  const float v = qkv[base + 2 * NC + h * NHD + d];

  red[d] = q * q; __syncthreads();
  for (int s = 32; s > 0; s >>= 1) { if (d < s) red[d] += red[d + s]; __syncthreads(); }
  const float nq = sqrtf(red[0]); __syncthreads();
  red[d] = k * k; __syncthreads();
  for (int s = 32; s > 0; s >>= 1) { if (d < s) red[d] += red[d + s]; __syncthreads(); }
  const float nk = sqrtf(red[0]);

  const float smul = __expf(fminf(slog[h], MAX_SCALEF));
  const float qn = q / fmaxf(nq, 1e-12f) * smul;
  const float kn = k / fmaxf(nk, 1e-12f);

  const float c = fcos[l * (NHD / 2) + (d >> 1)];
  const float s = fsin[l * (NHD / 2) + (d >> 1)];
  const float qp = __shfl_xor(qn, 1, 32);
  const float kp = __shfl_xor(kn, 1, 32);
  const float qr = (d & 1) ? (qp * s + qn * c) : (qn * c - qp * s);
  const float kr = (d & 1) ? (kp * s + kn * c) : (kn * c - kp * s);

  const size_t bh = (size_t)(b * NH + h);
  qb[(bh * NL + l) * NHD + d] = f2bf(qr);
  kb[(bh * NL + l) * NHD + d] = f2bf(kr);
  vt[(bh * NHD + d) * NL + l] = f2bf(v);
}

// ---------- tiled WMMA GEMM: out = A(bf16,MxK) @ W(bf16,[N,K])^T + bias ----------
// Block tile 128x64, 8 waves, each wave 32x32 (4 WMMAs / K-step), double-buffered LDS.
// EPI 0: f32 out ; EPI 1: gelu -> bf16 out
template <int EPI>
__global__ __launch_bounds__(256)
void k_gemm(const unsigned short* __restrict__ A, const unsigned short* __restrict__ W,
            const float* __restrict__ bias, void* __restrict__ out,
            int Nn, int Kk) {
  __shared__ __align__(16) unsigned short As[2][128 * 32];
  __shared__ __align__(16) unsigned short Bs[2][64 * 32];
  const int t = threadIdx.x, lane = t & 31, wave = t >> 5;
  const int wm = wave & 3, wn = wave >> 2;
  const int r16 = lane & 15, sel = lane >> 4, kb8 = sel * 8;
  const int m0 = blockIdx.y * 128, n0 = blockIdx.x * 64;

  // cooperative-load mapping (v8bf granules)
  const int ldr = t >> 2, ldc = (t & 3) * 8;        // rows 0..63, col {0,8,16,24}
  const size_t aoff0 = (size_t)(m0 + ldr) * Kk + ldc;
  const size_t aoff1 = (size_t)(m0 + 64 + ldr) * Kk + ldc;
  const size_t boff  = (size_t)(n0 + ldr) * Kk + ldc;

  const v8f vz = {0.f,0.f,0.f,0.f,0.f,0.f,0.f,0.f};
  v8f acc00 = vz, acc01 = vz, acc10 = vz, acc11 = vz;

  v8bf ra0 = *(const v8bf*)&A[aoff0];
  v8bf ra1 = *(const v8bf*)&A[aoff1];
  v8bf rb  = *(const v8bf*)&W[boff];

  int buf = 0;
  for (int k0 = 0; k0 < Kk; k0 += 32) {
    *(v8bf*)&As[buf][ldr * 32 + ldc]        = ra0;
    *(v8bf*)&As[buf][(64 + ldr) * 32 + ldc] = ra1;
    *(v8bf*)&Bs[buf][ldr * 32 + ldc]        = rb;
    __syncthreads();
    if (k0 + 32 < Kk) {                       // prefetch next K-slab into registers
      ra0 = *(const v8bf*)&A[aoff0 + k0 + 32];
      ra1 = *(const v8bf*)&A[aoff1 + k0 + 32];
      rb  = *(const v8bf*)&W[boff + k0 + 32];
    }
    const unsigned short* as = &As[buf][0];
    const unsigned short* bs = &Bs[buf][0];
    // batch all 8 LDS fragment loads, then 4 WMMAs
    const v8bf a0l = *(const v8bf*)&as[(wm * 32 + r16) * 32 + kb8];
    const v8bf a0h = *(const v8bf*)&as[(wm * 32 + r16) * 32 + kb8 + 16];
    const v8bf a1l = *(const v8bf*)&as[(wm * 32 + 16 + r16) * 32 + kb8];
    const v8bf a1h = *(const v8bf*)&as[(wm * 32 + 16 + r16) * 32 + kb8 + 16];
    const v8bf b0l = *(const v8bf*)&bs[(wn * 32 + r16) * 32 + kb8];
    const v8bf b0h = *(const v8bf*)&bs[(wn * 32 + r16) * 32 + kb8 + 16];
    const v8bf b1l = *(const v8bf*)&bs[(wn * 32 + 16 + r16) * 32 + kb8];
    const v8bf b1h = *(const v8bf*)&bs[(wn * 32 + 16 + r16) * 32 + kb8 + 16];
    const v16bf a0 = cat16(a0l, a0h), a1 = cat16(a1l, a1h);
    const v16bf b0 = cat16(b0l, b0h), b1 = cat16(b1l, b1h);
    acc00 = WMMA_BF16(a0, b0, acc00);
    acc01 = WMMA_BF16(a0, b1, acc01);
    acc10 = WMMA_BF16(a1, b0, acc10);
    acc11 = WMMA_BF16(a1, b1, acc11);
    buf ^= 1;
  }

  const int n0a = n0 + wn * 32 + r16;
  const int n1a = n0a + 16;
  const float bz0 = bias[n0a], bz1 = bias[n1a];
#pragma unroll
  for (int r = 0; r < 8; ++r) {
    const int ma = m0 + wm * 32 + sel * 8 + r;
    const int mb = ma + 16;
    const float v00 = acc00[r] + bz0, v01 = acc01[r] + bz1;
    const float v10 = acc10[r] + bz0, v11 = acc11[r] + bz1;
    if constexpr (EPI == 0) {
      ((float*)out)[(size_t)ma * Nn + n0a] = v00;
      ((float*)out)[(size_t)ma * Nn + n1a] = v01;
      ((float*)out)[(size_t)mb * Nn + n0a] = v10;
      ((float*)out)[(size_t)mb * Nn + n1a] = v11;
    } else {
      ((unsigned short*)out)[(size_t)ma * Nn + n0a] = f2bf(gelu_t(v00));
      ((unsigned short*)out)[(size_t)ma * Nn + n1a] = f2bf(gelu_t(v01));
      ((unsigned short*)out)[(size_t)mb * Nn + n0a] = f2bf(gelu_t(v10));
      ((unsigned short*)out)[(size_t)mb * Nn + n1a] = f2bf(gelu_t(v11));
    }
  }
}

// ---------------- flash attention: 4 waves x 16 q rows, 32-key tiles, double-buffered ----------------
__global__ __launch_bounds__(128)
void k_attn(const unsigned short* __restrict__ qbuf, const unsigned short* __restrict__ kbuf,
            const unsigned short* __restrict__ vtbuf, const float* __restrict__ abias,
            unsigned short* __restrict__ obuf) {
  __shared__ __align__(16) unsigned short Ks[2][32 * 64];   // [key][d]
  __shared__ __align__(16) unsigned short Vts[2][64 * 32];  // [d][key]
  __shared__ __align__(16) unsigned short Ps[4 * 16 * 32];  // per-wave P tile
  const int t = threadIdx.x, lane = t & 31, wave = t >> 5;
  const int r16 = lane & 15, sel = lane >> 4, kb8 = sel * 8;
  const int bh = blockIdx.x, b = bh >> 4, h = bh & 15;
  const int q0 = blockIdx.y * 64;

  const unsigned short* qrowp = qbuf + ((size_t)bh * NL + q0 + wave * 16 + r16) * NHD;
  const v16bf qf0 = cat16(*(const v8bf*)&qrowp[kb8],      *(const v8bf*)&qrowp[kb8 + 16]);
  const v16bf qf1 = cat16(*(const v8bf*)&qrowp[32 + kb8], *(const v8bf*)&qrowp[32 + kb8 + 16]);

  const v8f vz = {0.f,0.f,0.f,0.f,0.f,0.f,0.f,0.f};
  v8f o0 = vz, o1 = vz, o2 = vz, o3 = vz;
  float mrow[8], lrow[8];
#pragma unroll
  for (int r = 0; r < 8; ++r) { mrow[r] = -3.0e38f; lrow[r] = 0.f; }

  // cooperative-load mapping + register staging
  const int kkey = t >> 2, kdd = (t & 3) * 16;
  const int vdD = t >> 1, vcs = (t & 1) * 16;
  const unsigned short* ksrc = kbuf + ((size_t)bh * NL + kkey) * NHD + kdd;
  const unsigned short* vsrc = vtbuf + ((size_t)bh * NHD + vdD) * NL + vcs;
  v8bf rk0 = *(const v8bf*)&ksrc[0];
  v8bf rk1 = *(const v8bf*)&ksrc[8];
  v8bf rv0 = *(const v8bf*)&vsrc[0];
  v8bf rv1 = *(const v8bf*)&vsrc[8];

  int buf = 0;
  for (int kv0 = 0; kv0 < NL; kv0 += 32) {
    *(v8bf*)&Ks[buf][kkey * 64 + kdd]      = rk0;
    *(v8bf*)&Ks[buf][kkey * 64 + kdd + 8]  = rk1;
    *(v8bf*)&Vts[buf][vdD * 32 + vcs]      = rv0;
    *(v8bf*)&Vts[buf][vdD * 32 + vcs + 8]  = rv1;
    __syncthreads();
    if (kv0 + 32 < NL) {                    // prefetch next tile into registers
      rk0 = *(const v8bf*)&ksrc[(size_t)(kv0 + 32) * NHD];
      rk1 = *(const v8bf*)&ksrc[(size_t)(kv0 + 32) * NHD + 8];
      rv0 = *(const v8bf*)&vsrc[kv0 + 32];
      rv1 = *(const v8bf*)&vsrc[kv0 + 32 + 8];
    }
    const unsigned short* ks = &Ks[buf][0];
    const unsigned short* vs = &Vts[buf][0];

    // batch all 8 K-fragment LDS loads, then 4 WMMAs
    const v8bf k00l = *(const v8bf*)&ks[r16 * 64 + kb8];
    const v8bf k00h = *(const v8bf*)&ks[r16 * 64 + kb8 + 16];
    const v8bf k01l = *(const v8bf*)&ks[r16 * 64 + 32 + kb8];
    const v8bf k01h = *(const v8bf*)&ks[r16 * 64 + 32 + kb8 + 16];
    const v8bf k10l = *(const v8bf*)&ks[(16 + r16) * 64 + kb8];
    const v8bf k10h = *(const v8bf*)&ks[(16 + r16) * 64 + kb8 + 16];
    const v8bf k11l = *(const v8bf*)&ks[(16 + r16) * 64 + 32 + kb8];
    const v8bf k11h = *(const v8bf*)&ks[(16 + r16) * 64 + 32 + kb8 + 16];
    const v16bf k00 = cat16(k00l, k00h), k01 = cat16(k01l, k01h);
    const v16bf k10 = cat16(k10l, k10h), k11 = cat16(k11l, k11h);
    v8f S0 = vz, S1 = vz;
    S0 = WMMA_BF16(qf0, k00, S0);
    S1 = WMMA_BF16(qf0, k10, S1);
    S0 = WMMA_BF16(qf1, k01, S0);
    S1 = WMMA_BF16(qf1, k11, S1);

#pragma unroll
    for (int r = 0; r < 8; ++r) {
      const int qrow = q0 + wave * 16 + sel * 8 + r;
      float s0 = S0[r] + abias[(size_t)qrow * NL + kv0 + r16];
      float s1 = S1[r] + abias[(size_t)qrow * NL + kv0 + 16 + r16];
      const float mx = red16_max(fmaxf(s0, s1));
      const float mnew = fmaxf(mrow[r], mx);
      const float fac = __expf(mrow[r] - mnew);
      const float p0 = __expf(s0 - mnew), p1 = __expf(s1 - mnew);
      const float rs = red16_sum(p0 + p1);
      lrow[r] = lrow[r] * fac + rs;
      mrow[r] = mnew;
      o0[r] *= fac; o1[r] *= fac; o2[r] *= fac; o3[r] *= fac;
      Ps[wave * 512 + (sel * 8 + r) * 32 + r16]      = f2bf(p0);
      Ps[wave * 512 + (sel * 8 + r) * 32 + 16 + r16] = f2bf(p1);
    }
    asm volatile("s_wait_dscnt 0x0" ::: "memory");

    // batch all P/V fragment LDS loads, then 4 WMMAs
    const v8bf pfl = *(const v8bf*)&Ps[wave * 512 + r16 * 32 + kb8];
    const v8bf pfh = *(const v8bf*)&Ps[wave * 512 + r16 * 32 + kb8 + 16];
    const v8bf v0l = *(const v8bf*)&vs[(0 * 16 + r16) * 32 + kb8];
    const v8bf v0h = *(const v8bf*)&vs[(0 * 16 + r16) * 32 + kb8 + 16];
    const v8bf v1l = *(const v8bf*)&vs[(1 * 16 + r16) * 32 + kb8];
    const v8bf v1h = *(const v8bf*)&vs[(1 * 16 + r16) * 32 + kb8 + 16];
    const v8bf v2l = *(const v8bf*)&vs[(2 * 16 + r16) * 32 + kb8];
    const v8bf v2h = *(const v8bf*)&vs[(2 * 16 + r16) * 32 + kb8 + 16];
    const v8bf v3l = *(const v8bf*)&vs[(3 * 16 + r16) * 32 + kb8];
    const v8bf v3h = *(const v8bf*)&vs[(3 * 16 + r16) * 32 + kb8 + 16];
    const v16bf pf  = cat16(pfl, pfh);
    const v16bf vf0 = cat16(v0l, v0h), vf1 = cat16(v1l, v1h);
    const v16bf vf2 = cat16(v2l, v2h), vf3 = cat16(v3l, v3h);
    o0 = WMMA_BF16(pf, vf0, o0);
    o1 = WMMA_BF16(pf, vf1, o1);
    o2 = WMMA_BF16(pf, vf2, o2);
    o3 = WMMA_BF16(pf, vf3, o3);
    buf ^= 1;
  }

#pragma unroll
  for (int r = 0; r < 8; ++r) {
    const int qrow = q0 + wave * 16 + sel * 8 + r;
    const float inv = 1.f / lrow[r];
    const size_t ob = ((size_t)b * NL + qrow) * NC + h * NHD;
    obuf[ob + r16]      = f2bf(o0[r] * inv);
    obuf[ob + 16 + r16] = f2bf(o1[r] * inv);
    obuf[ob + 32 + r16] = f2bf(o2[r] * inv);
    obuf[ob + 48 + r16] = f2bf(o3[r] * inv);
  }
}

// ---------------- host ----------------

extern "C" void kernel_launch(void* const* d_in, const int* in_sizes, int n_in,
                              void* d_out, int out_size, void* d_ws, size_t ws_size,
                              hipStream_t stream) {
  const float* x     = (const float*)d_in[0];
  const float* cond  = (const float*)d_in[1];
  const float* fcos  = (const float*)d_in[2];
  const float* fsin  = (const float*)d_in[3];
  const float* abias = (const float*)d_in[4];
  const float* Wqkv  = (const float*)d_in[5];
  const float* qbias = (const float*)d_in[6];
  const float* vbias = (const float*)d_in[7];
  const float* slog  = (const float*)d_in[8];
  const float* Wproj = (const float*)d_in[9];
  const float* bproj = (const float*)d_in[10];
  const float* Wfc1  = (const float*)d_in[11];
  const float* bfc1  = (const float*)d_in[12];
  const float* Wfc2  = (const float*)d_in[13];
  const float* bfc2  = (const float*)d_in[14];
  const float* Wada  = (const float*)d_in[15];
  const float* bada  = (const float*)d_in[16];
  float* out = (float*)d_out;

  char* ws = (char*)d_ws;
  size_t off = 0;
  auto alloc = [&](size_t bytes) -> char* {
    char* p = ws + off;
    off += (bytes + 255) & ~(size_t)255;
    return p;
  };
  unsigned short* wqkv_b  = (unsigned short*)alloc((size_t)3 * NC * NC * 2);
  unsigned short* wproj_b = (unsigned short*)alloc((size_t)NC * NC * 2);
  unsigned short* wfc1_b  = (unsigned short*)alloc((size_t)4 * NC * NC * 2);
  unsigned short* wfc2_b  = (unsigned short*)alloc((size_t)4 * NC * NC * 2);
  float* bqkv = (float*)alloc((size_t)3 * NC * 4);
  float* ada  = (float*)alloc((size_t)NB * 6 * NC * 4);
  unsigned short* hbuf = (unsigned short*)alloc((size_t)NM * NC * 2);
  unsigned short* qb   = (unsigned short*)alloc((size_t)NM * NC * 2);
  unsigned short* kbf  = (unsigned short*)alloc((size_t)NM * NC * 2);
  unsigned short* vt   = (unsigned short*)alloc((size_t)NM * NC * 2);
  unsigned short* atto = (unsigned short*)alloc((size_t)NM * NC * 2);
  float* x2  = (float*)alloc((size_t)NM * NC * 4);
  float* big = (float*)alloc((size_t)NM * 3 * NC * 4);   // qkv / proj / fc2 (sequential lifetimes)
  unsigned short* gelu_b = qb;  // alias: q/k/v/attn-out region (exactly 67 MB) is dead by FC1

  // weight conversion + bias concat + adaLN params
  k_cvt<<<(3 * NC * NC + 255) / 256, 256, 0, stream>>>(Wqkv, wqkv_b, 3 * NC * NC);
  k_cvt<<<(NC * NC + 255) / 256, 256, 0, stream>>>(Wproj, wproj_b, NC * NC);
  k_cvt<<<(4 * NC * NC + 255) / 256, 256, 0, stream>>>(Wfc1, wfc1_b, 4 * NC * NC);
  k_cvt<<<(4 * NC * NC + 255) / 256, 256, 0, stream>>>(Wfc2, wfc2_b, 4 * NC * NC);
  k_qkvbias<<<12, 256, 0, stream>>>(qbias, vbias, bqkv);
  k_ada<<<NB * 24, 256, 0, stream>>>(cond, Wada, bada, ada);

  // attention branch
  k_ln_mod<<<NM, 256, 0, stream>>>(x, ada, 2, 4, hbuf);
  k_gemm<0><<<dim3(48, 64), 256, 0, stream>>>(hbuf, wqkv_b, bqkv, big, 3 * NC, NC);
  k_qkv_prep<<<NB * NL * NH, 64, 0, stream>>>(big, slog, fcos, fsin, qb, kbf, vt);
  k_attn<<<dim3(NB * NH, NL / 64), 128, 0, stream>>>(qb, kbf, vt, abias, atto);
  k_gemm<0><<<dim3(16, 64), 256, 0, stream>>>(atto, wproj_b, bproj, big, NC, NC);
  k_resid<<<(NM * NC) / 256, 256, 0, stream>>>(x, big, ada, 0, x2);

  // FFN branch
  k_ln_mod<<<NM, 256, 0, stream>>>(x2, ada, 3, 5, hbuf);
  k_gemm<1><<<dim3(64, 64), 256, 0, stream>>>(hbuf, wfc1_b, bfc1, gelu_b, 4 * NC, NC);
  k_gemm<0><<<dim3(16, 64), 256, 0, stream>>>(gelu_b, wfc2_b, bfc2, big, NC, 4 * NC);
  k_resid<<<(NM * NC) / 256, 256, 0, stream>>>(x2, big, ada, 1, out);

  (void)in_sizes; (void)n_in; (void)out_size; (void)ws_size;
}
